// MoeGate_17867063951952
// MI455X (gfx1250) — matrix operand order repaired and checked by reading, not compiled, not run on hardware
//
#include <hip/hip_runtime.h>
#include <hip/hip_bf16.h>
#include <stdint.h>

typedef __attribute__((ext_vector_type(16))) __bf16 v16bf;
typedef __attribute__((ext_vector_type(8)))  float  v8f;

constexpr int kDim    = 7168;
constexpr int kNE     = 256;     // experts
constexpr int kNT     = 16384;   // tokens
constexpr int kTopK   = 8;
constexpr int kNG     = 8;       // groups
constexpr int kTopG   = 4;
constexpr int kEPG    = 32;      // experts per group
constexpr int kMTile  = 32;      // tokens per block
constexpr int kKChunk = 896;     // K chunk staged in LDS (7168 = 8 * 896)
constexpr int kAPad   = 904;     // 896 + 8 halves pad -> conflict-free ds_load_b128
constexpr int kSPad   = 260;     // score row stride (floats)
constexpr int kSmemB  = kMTile * kAPad * 2;          // 57,856 B (A tile dominates)
constexpr int kSGOff  = kMTile * kSPad * 4;          // 33,280 B (scores), sG after

union FragBits { uint4 u[2]; v16bf v; };

static __device__ __forceinline__ unsigned pk_bf16(float lo, float hi) {
  __bf16 l = (__bf16)lo;
  __bf16 h = (__bf16)hi;
  unsigned short ls = __builtin_bit_cast(unsigned short, l);
  unsigned short hs = __builtin_bit_cast(unsigned short, h);
  return (unsigned)ls | ((unsigned)hs << 16);
}

// One-shot fp32 -> bf16 conversion of the 256x7168 expert matrix into workspace.
// 3.67 MB result stays L2-resident for the GEMM kernel.
__global__ __launch_bounds__(256)
void convert_w_kernel(const float* __restrict__ w, __bf16* __restrict__ wbf) {
  const size_t i = (size_t)blockIdx.x * 256 + threadIdx.x;   // float4 index
  float4 f = ((const float4*)w)[i];
  uint2 p;
  p.x = pk_bf16(f.x, f.y);
  p.y = pk_bf16(f.z, f.w);
  ((uint2*)wbf)[i] = p;
}

__global__ __launch_bounds__(256)
void moe_gate_kernel(const float* __restrict__ x,
                     const __bf16* __restrict__ wbf,
                     const float* __restrict__ bias,
                     float* __restrict__ out_w,
                     int*   __restrict__ out_i) {
  // One shared buffer, two lifetimes:
  //   phase 1 (GEMM): bf16 A tile  [32][904]          = 57,856 B
  //   phase 2 (topk): fp32 scores  [32][260] + sG[256] = 34,304 B (aliases A)
  __shared__ __align__(16) char smem[kSmemB];
  __bf16* sA = (__bf16*)smem;
  float*  sS = (float*)smem;
  float*  sG = (float*)(smem + kSGOff);

  const int tid  = threadIdx.x;
  const int lane = tid & 31;
  const int wave = tid >> 5;       // 0..7, each owns 32 experts
  const int lrow = lane & 15;
  const int lhi  = lane >> 4;      // 0 or 1
  const int row0 = blockIdx.x * kMTile;
  const int nb0  = wave * 32;

  v8f acc00 = {}, acc01 = {};      // token rows 0..15  x experts nb0+0..15 / +16..31
  v8f acc10 = {}, acc11 = {};      // token rows 16..31 x experts nb0+0..15 / +16..31

  // Staging map: 8 threads per token row, 28 float4s each (896 floats/row).
  const int srow = tid >> 3;       // 0..31
  const int sj   = tid & 7;        // 0..7

  for (int kc = 0; kc < kDim; kc += kKChunk) {
    __syncthreads();
    {
      const float4* xg = (const float4*)(x + (size_t)(row0 + srow) * kDim + kc);
      #pragma unroll
      for (int i = 0; i < 28; ++i) {
        const int c4 = sj + 8 * i;           // float4 column within chunk
        float4 f = xg[c4];
        uint2 p;
        p.x = pk_bf16(f.x, f.y);
        p.y = pk_bf16(f.z, f.w);
        *(uint2*)(sA + srow * kAPad + c4 * 4) = p;
      }
    }
    __syncthreads();

    // bf16 W row bases for this wave's two 16-expert fragments (L2-resident).
    // B layout: column = nb + (lane&15); lanes 0-15 hold K+0..15, lanes 16-31 hold K+16..31.
    const __bf16* wb0 = wbf + (size_t)(nb0 + lrow)      * kDim + kc + lhi * 16;
    const __bf16* wb1 = wbf + (size_t)(nb0 + 16 + lrow) * kDim + kc + lhi * 16;
    // A fragment bases in LDS for the two 16-token tiles.
    const __bf16* ar0 = sA + lrow * kAPad + lhi * 16;
    const __bf16* ar1 = ar0 + 16 * kAPad;

    #pragma unroll 2
    for (int kk = 0; kk < kKChunk; kk += 32) {
      FragBits a0, a1, b0, b1;
      a0.u[0] = *(const uint4*)(ar0 + kk);
      a0.u[1] = *(const uint4*)(ar0 + kk + 8);
      a1.u[0] = *(const uint4*)(ar1 + kk);
      a1.u[1] = *(const uint4*)(ar1 + kk + 8);
      b0.u[0] = *(const uint4*)(wb0 + kk);
      b0.u[1] = *(const uint4*)(wb0 + kk + 8);
      b1.u[0] = *(const uint4*)(wb1 + kk);
      b1.u[1] = *(const uint4*)(wb1 + kk + 8);

      acc00 = __builtin_amdgcn_wmma_f32_16x16x32_bf16(false, a0.v, false, b0.v,
                                                      (short)0, acc00, false, false);
      acc01 = __builtin_amdgcn_wmma_f32_16x16x32_bf16(false, a0.v, false, b1.v,
                                                      (short)0, acc01, false, false);
      acc10 = __builtin_amdgcn_wmma_f32_16x16x32_bf16(false, a1.v, false, b0.v,
                                                      (short)0, acc10, false, false);
      acc11 = __builtin_amdgcn_wmma_f32_16x16x32_bf16(false, a1.v, false, b1.v,
                                                      (short)0, acc11, false, false);
    }
  }

  __syncthreads();   // last read of sA done; smem becomes the score buffer
  // C/D layout: VGPR j -> M=j (lanes 0-15) / M=j+8 (lanes 16-31), N = nb + (lane&15).
  #pragma unroll
  for (int j = 0; j < 8; ++j) {
    const int m0 = j + lhi * 8;
    const int m1 = 16 + m0;
    sS[m0 * kSPad + nb0 + lrow]      = __fdividef(1.0f, 1.0f + __expf(-acc00[j]));
    sS[m0 * kSPad + nb0 + 16 + lrow] = __fdividef(1.0f, 1.0f + __expf(-acc01[j]));
    sS[m1 * kSPad + nb0 + lrow]      = __fdividef(1.0f, 1.0f + __expf(-acc10[j]));
    sS[m1 * kSPad + nb0 + 16 + lrow] = __fdividef(1.0f, 1.0f + __expf(-acc11[j]));
  }
  __syncthreads();

  // Group criterion: sum of top-2 corrected scores per group (8 threads / token; 32*8 = 256).
  {
    const int tok = tid >> 3, g = tid & 7;
    const float* sr = sS + tok * kSPad + g * kEPG;
    const float* br = bias + g * kEPG;
    float t1 = -INFINITY, t2 = -INFINITY;
    for (int e = 0; e < kEPG; ++e) {
      const float c = sr[e] + br[e];
      if (c > t1)      { t2 = t1; t1 = c; }
      else if (c > t2) { t2 = c; }
    }
    sG[tok * kNG + g] = t1 + t2;
  }
  __syncthreads();

  // Final selection: one thread per token.
  if (tid < kMTile) {
    const int tok = tid;
    float gsc[kNG];
    for (int g = 0; g < kNG; ++g) gsc[g] = sG[tok * kNG + g];

    unsigned gmask = 0;
    for (int s = 0; s < kTopG; ++s) {
      int bi = 0; float bv = -INFINITY;
      for (int g = 0; g < kNG; ++g)
        if (!(gmask & (1u << g)) && gsc[g] > bv) { bv = gsc[g]; bi = g; }
      gmask |= 1u << bi;
    }

    float val[kTopK]; int idx[kTopK];
    for (int i = 0; i < kTopK; ++i) { val[i] = -INFINITY; idx[i] = -1; }
    const float* sr = sS + tok * kSPad;
    for (int e = 0; e < kNE; ++e) {
      if (!(gmask & (1u << (e >> 5)))) continue;
      const float c = sr[e] + bias[e];
      if (c > val[kTopK - 1]) {
        int j = kTopK - 1;
        while (j > 0 && c > val[j - 1]) { val[j] = val[j - 1]; idx[j] = idx[j - 1]; --j; }
        val[j] = c; idx[j] = e;
      }
    }

    float sum = 0.0f, wsel[kTopK];
    for (int i = 0; i < kTopK; ++i) { wsel[i] = sr[idx[i]]; sum += wsel[i]; }
    const float inv = 2.5f / sum;
    const int tg = row0 + tok;
    for (int i = 0; i < kTopK; ++i) {
      out_w[tg * kTopK + i] = wsel[i] * inv;
      out_i[tg * kTopK + i] = idx[i];
    }
  }
}

extern "C" void kernel_launch(void* const* d_in, const int* in_sizes, int n_in,
                              void* d_out, int out_size, void* d_ws, size_t ws_size,
                              hipStream_t stream) {
  (void)in_sizes; (void)n_in; (void)out_size; (void)ws_size;
  const float* x  = (const float*)d_in[0];
  const float* w  = (const float*)d_in[1];
  const float* cb = (const float*)d_in[2];
  float* out_w = (float*)d_out;
  int*   out_i = (int*)((float*)d_out + (size_t)kNT * kTopK);
  __bf16* wbf  = (__bf16*)d_ws;   // 256*7168*2 = 3.67 MB scratch

  // Pass 1: W fp32 -> bf16 (one float4 per thread).
  {
    dim3 grid((kNE * kDim / 4) / 256), block(256);
    hipLaunchKernelGGL(convert_w_kernel, grid, block, 0, stream, w, wbf);
  }
  // Pass 2: fused gated-GEMM + grouped top-k routing.
  {
    dim3 grid(kNT / kMTile), block(256);
    hipLaunchKernelGGL(moe_gate_kernel, grid, block, 0, stream, x, wbf, cb, out_w, out_i);
  }
}
